// GCN_23261542875425
// MI455X (gfx1250) — matrix operand order repaired and checked by reading, not compiled, not run on hardware
//
#include <hip/hip_runtime.h>

// ---------------------------------------------------------------------------
// GCN pipeline for MI455X (gfx1250, wave32, WMMA + async global->LDS).
// N=100000 nodes, E=1600000 edges, C=128/128/64, 64 graphs, 2 classes.
// ---------------------------------------------------------------------------

#define N_NODES 100000
#define N_EDGES 1600000
#define N_GRAPHS 64
#define IN_CH 128
#define HID_CH 128
#define OUT_CH 64
#define N_CLS 2

typedef __attribute__((ext_vector_type(16))) __bf16 v16bf;
typedef __attribute__((ext_vector_type(8)))  float  v8f;

union BFrag {
    v16bf v;
    unsigned short s[16];
    uint4 q[2];
};

static __device__ __forceinline__ unsigned short f32_to_bf16_bits(float f) {
    union { float f; unsigned u; } x;
    x.f = f;
    unsigned r = x.u + 0x7FFFu + ((x.u >> 16) & 1u);   // round-to-nearest-even
    return (unsigned short)(r >> 16);
}

// ---------------------------------------------------------------------------
// Degree helpers
// ---------------------------------------------------------------------------
__global__ void fill_ones_kernel(float* __restrict__ p, int n) {
    int i = blockIdx.x * blockDim.x + threadIdx.x;
    if (i < n) p[i] = 1.0f;
}

__global__ void deg_count_kernel(const int* __restrict__ dst, float* __restrict__ deg, int nE) {
    int i = blockIdx.x * blockDim.x + threadIdx.x;
    if (i < nE)
        __hip_atomic_fetch_add(&deg[dst[i]], 1.0f, __ATOMIC_RELAXED, __HIP_MEMORY_SCOPE_AGENT);
}

__global__ void rsqrt_kernel(float* __restrict__ p, int n) {
    int i = blockIdx.x * blockDim.x + threadIdx.x;
    if (i < n) p[i] = rsqrtf(p[i]);
}

// ---------------------------------------------------------------------------
// GEMM:  H[nRows x NOUT] = act(A[nRows x 128]) @ W[128 x NOUT]
// bf16 WMMA (v_wmma_f32_16x16x32_bf16), f32 accumulate.
// Block = 256 threads = 8 waves; one block computes a 128-row strip.
//
// Full strips stage the 64 KB f32 A-strip with GLOBAL_LOAD_ASYNC_TO_LDS_B128
// (ASYNCcnt path), completed before W staging; then convert LDS f32 ->
// fragment-ordered bf16 LDS so each wave fetches a 16x32 fragment with two
// ds_load_b128 per lane.
//
// A fragment layout (16-bit A 16x32, ISA 7.12.2): lane = hl*16 + (row%16),
//   element e<8 -> K = kb*32 + hl*8 + e ; e>=8 -> K = kb*32 + 16 + hl*8 + (e-8)
// B fragment layout: lane = hl*16 + (col%16), element e -> K as in the A layout above
// C/D: VGPR r holds (M = r + hl*8, N = lane%16).
// ---------------------------------------------------------------------------
template <int NOUT, bool RELU_IN>
__global__ __launch_bounds__(256) void gemm_bf16_wmma_kernel(
        const float* __restrict__ A, const float* __restrict__ W,
        float* __restrict__ H, int nRows) {
    constexpr int K = 128;
    __shared__ float lXf32[128 * K];                     // 64 KB async staging
    __shared__ unsigned short lA[8][4][32][16];          // [rowtile][kb][lane][e]  32 KB
    __shared__ unsigned short lB[NOUT / 16][4][32][16];  // [coltile][kb][lane][e]  16/32 KB

    const int tid = threadIdx.x;
    const int r0 = blockIdx.x * 128;
    const bool fullStrip = (r0 + 128 <= nRows);

    // Stage A strip (f32) into LDS.
    if (fullStrip) {
        // gfx1250 async copy: per-lane 16B global->LDS, tracked by ASYNCcnt.
        const float* gbase = A + (size_t)r0 * K;
        const unsigned ldsBase = (unsigned)(size_t)(&lXf32[0]);
        #pragma unroll
        for (int j = 0; j < (128 * K * 4) / (256 * 16); ++j) {   // 16 iters
            unsigned byteOff = (unsigned)(tid + j * 256) * 16u;
            asm volatile("global_load_async_to_lds_b128 %0, %1, %2"
                         :: "v"(ldsBase + byteOff), "v"(byteOff), "s"(gbase)
                         : "memory");
        }
        asm volatile("s_wait_asynccnt 0x0" ::: "memory");
    } else {
        for (int i = tid; i < 128 * K; i += 256) {
            int gr = r0 + (i >> 7);
            lXf32[i] = (gr < nRows) ? A[(size_t)gr * K + (i & (K - 1))] : 0.0f;
        }
    }

    // Stage W -> fragment-ordered LDS (bf16).
    for (int i = tid; i < K * NOUT; i += 256) {
        int k = i / NOUT, n = i % NOUT;
        int ct = n >> 4, l = n & 15;
        int kb = k >> 5, kk = k & 31;
        int hl = (kk >> 3) & 1, e = (kk & 7) + ((kk >> 4) << 3);
        lB[ct][kb][hl * 16 + l][e] = f32_to_bf16_bits(W[i]);
    }
    __syncthreads();

    // Convert A strip: LDS f32 -> fragment-ordered bf16 (fused ReLU).
    for (int i = tid; i < 128 * K; i += 256) {
        float v = lXf32[i];
        if (RELU_IN) v = v > 0.0f ? v : 0.0f;
        int row = i >> 7, k = i & (K - 1);
        int rt = row >> 4, l = row & 15;
        int kb = k >> 5, kk = k & 31;
        int hl = (kk >> 3) & 1;
        int e = (kk & 7) + ((kk >> 4) << 3);
        lA[rt][kb][hl * 16 + l][e] = f32_to_bf16_bits(v);
    }
    __syncthreads();

    const int wave = tid >> 5;
    const int lane = tid & 31;
    const int hl = lane >> 4;
    const int l16 = lane & 15;

    // Hoist the wave's 4 A-fragments (reused across every column tile)
    BFrag a[4];
    #pragma unroll
    for (int kb = 0; kb < 4; ++kb) {
        const uint4* p = reinterpret_cast<const uint4*>(&lA[wave][kb][lane][0]);
        a[kb].q[0] = p[0];
        a[kb].q[1] = p[1];
    }

    #pragma unroll
    for (int ct = 0; ct < NOUT / 16; ++ct) {
        v8f acc = {};
        #pragma unroll
        for (int kb = 0; kb < 4; ++kb) {
            BFrag b;
            const uint4* p = reinterpret_cast<const uint4*>(&lB[ct][kb][lane][0]);
            b.q[0] = p[0];
            b.q[1] = p[1];
            acc = __builtin_amdgcn_wmma_f32_16x16x32_bf16(
                false, a[kb].v, false, b.v, (short)0, acc, false, false);
        }
        const int col = ct * 16 + l16;
        const int mBase = r0 + wave * 16 + hl * 8;
        if (fullStrip) {
            // Block-uniform fast path: 8 unguarded stores, immediate offsets.
            float* base = H + (size_t)mBase * NOUT + col;
            #pragma unroll
            for (int r = 0; r < 8; ++r) base[(size_t)r * NOUT] = acc[r];
        } else {
            #pragma unroll
            for (int r = 0; r < 8; ++r) {
                int gr = mBase + r;
                if (gr < nRows) H[(size_t)gr * NOUT + col] = acc[r];
            }
        }
    }
}

// ---------------------------------------------------------------------------
// agg[n][c] = h[n][c] * dinv[n]^2 + b[c]    (self-loop term + bias; this also
// initializes agg so the edge scatter needs no memset)
// ---------------------------------------------------------------------------
template <int C>
__global__ void init_agg_kernel(const float* __restrict__ h, const float* __restrict__ dinv,
                                const float* __restrict__ b, float* __restrict__ agg,
                                int total) {
    int i = blockIdx.x * blockDim.x + threadIdx.x;
    if (i >= total) return;
    int n = i / C;
    int c = i & (C - 1);
    float d = dinv[n];
    agg[i] = h[i] * d * d + b[c];
}

// ---------------------------------------------------------------------------
// Edge scatter: one wave per edge; agg[dst] += h[src] * dinv[src]*dinv[dst].
// float4 gathers (global_load_b128), native f32 atomics on the scatter side.
// ---------------------------------------------------------------------------
__global__ __launch_bounds__(256) void scatter_kernel(
        const int* __restrict__ src, const int* __restrict__ dst,
        const float* __restrict__ dinv, const float* __restrict__ h,
        float* __restrict__ agg, int nEdges, int C) {
    int e = (blockIdx.x * 256 + threadIdx.x) >> 5;
    if (e >= nEdges) return;
    int lane = threadIdx.x & 31;
    int s = src[e], d = dst[e];
    float w = dinv[s] * dinv[d];
    const float* hs = h + (size_t)s * C;
    float* ad = agg + (size_t)d * C;
    for (int c0 = lane * 4; c0 < C; c0 += 128) {
        float4 v = *reinterpret_cast<const float4*>(hs + c0);
        __hip_atomic_fetch_add(&ad[c0 + 0], v.x * w, __ATOMIC_RELAXED, __HIP_MEMORY_SCOPE_AGENT);
        __hip_atomic_fetch_add(&ad[c0 + 1], v.y * w, __ATOMIC_RELAXED, __HIP_MEMORY_SCOPE_AGENT);
        __hip_atomic_fetch_add(&ad[c0 + 2], v.z * w, __ATOMIC_RELAXED, __HIP_MEMORY_SCOPE_AGENT);
        __hip_atomic_fetch_add(&ad[c0 + 3], v.w * w, __ATOMIC_RELAXED, __HIP_MEMORY_SCOPE_AGENT);
    }
}

// ---------------------------------------------------------------------------
// Pooling: gsum[g][c] += agg3[n][c], cnt[g] += 1
// ---------------------------------------------------------------------------
__global__ void zero_kernel(float* __restrict__ p, int n) {
    int i = blockIdx.x * blockDim.x + threadIdx.x;
    if (i < n) p[i] = 0.0f;
}

__global__ void pool_accum_kernel(const float* __restrict__ agg, const int* __restrict__ batch,
                                  float* __restrict__ gsum, float* __restrict__ cnt, int nNodes) {
    int i = blockIdx.x * blockDim.x + threadIdx.x;
    if (i >= nNodes * OUT_CH) return;
    int n = i >> 6;
    int c = i & 63;
    int g = batch[n];
    __hip_atomic_fetch_add(&gsum[g * OUT_CH + c], agg[i], __ATOMIC_RELAXED, __HIP_MEMORY_SCOPE_AGENT);
    if (c == 0)
        __hip_atomic_fetch_add(&cnt[g], 1.0f, __ATOMIC_RELAXED, __HIP_MEMORY_SCOPE_AGENT);
}

// out[g][c] = (gsum[g]/max(cnt,1)) @ Wfc + bfc   -- 64x2, single block
__global__ void head_kernel(const float* __restrict__ gsum, const float* __restrict__ cnt,
                            const float* __restrict__ Wfc, const float* __restrict__ bfc,
                            float* __restrict__ out) {
    int t = threadIdx.x;
    if (t >= N_GRAPHS * N_CLS) return;
    int g = t >> 1, c = t & 1;
    float cn = cnt[g];
    cn = cn > 1.0f ? cn : 1.0f;
    float inv = 1.0f / cn;
    float s = 0.0f;
    for (int k = 0; k < OUT_CH; ++k)
        s += gsum[g * OUT_CH + k] * inv * Wfc[k * N_CLS + c];
    out[t] = s + bfc[c];
}

// ---------------------------------------------------------------------------
extern "C" void kernel_launch(void* const* d_in, const int* in_sizes, int n_in,
                              void* d_out, int out_size, void* d_ws, size_t ws_size,
                              hipStream_t stream) {
    const float* x    = (const float*)d_in[0];
    const int*   eidx = (const int*)d_in[1];
    const int*   src  = eidx;
    const int*   dst  = eidx + N_EDGES;
    const int*   batch = (const int*)d_in[2];
    const float* W1 = (const float*)d_in[3];
    const float* b1 = (const float*)d_in[4];
    const float* W2 = (const float*)d_in[5];
    const float* b2 = (const float*)d_in[6];
    const float* W3 = (const float*)d_in[7];
    const float* b3 = (const float*)d_in[8];
    const float* Wfc = (const float*)d_in[9];
    const float* bfc = (const float*)d_in[10];
    float* out = (float*)d_out;

    // Workspace layout (floats): dinv | h | agg | gsum | cnt
    float* ws   = (float*)d_ws;
    float* dinv = ws;                                   // N_NODES
    float* h    = ws + 100032;                          // N_NODES*128
    float* agg  = h + (size_t)N_NODES * 128;            // N_NODES*128
    float* gsum = agg + (size_t)N_NODES * 128;          // 64*64
    float* cnt  = gsum + N_GRAPHS * OUT_CH;             // 64 (contiguous after gsum)

    const int TB = 256;
    const int nBlkN    = (N_NODES + TB - 1) / TB;
    const int nBlkE    = (N_EDGES + TB - 1) / TB;
    const int nBlkGemm = (N_NODES + 127) / 128;         // 782 strips
    const int nBlk128  = (int)(((size_t)N_NODES * 128 + TB - 1) / TB);
    const int nBlk64   = (int)(((size_t)N_NODES * 64 + TB - 1) / TB);
    const int nBlkScat = (N_EDGES * 32 + TB - 1) / TB;  // one wave per edge

    // degrees: deg = 1 + indeg(dst); dinv = rsqrt(deg)
    fill_ones_kernel<<<nBlkN, TB, 0, stream>>>(dinv, N_NODES);
    deg_count_kernel<<<nBlkE, TB, 0, stream>>>(dst, dinv, N_EDGES);
    rsqrt_kernel<<<nBlkN, TB, 0, stream>>>(dinv, N_NODES);

    // Layer 1: h = x @ W1 ; agg = scatter + self + b1
    gemm_bf16_wmma_kernel<HID_CH, false><<<nBlkGemm, TB, 0, stream>>>(x, W1, h, N_NODES);
    init_agg_kernel<HID_CH><<<nBlk128, TB, 0, stream>>>(h, dinv, b1, agg, N_NODES * 128);
    scatter_kernel<<<nBlkScat, TB, 0, stream>>>(src, dst, dinv, h, agg, N_EDGES, HID_CH);

    // Layer 2: h = relu(agg) @ W2 (ReLU fused into GEMM staging)
    gemm_bf16_wmma_kernel<HID_CH, true><<<nBlkGemm, TB, 0, stream>>>(agg, W2, h, N_NODES);
    init_agg_kernel<HID_CH><<<nBlk128, TB, 0, stream>>>(h, dinv, b2, agg, N_NODES * 128);
    scatter_kernel<<<nBlkScat, TB, 0, stream>>>(src, dst, dinv, h, agg, N_EDGES, HID_CH);

    // Layer 3: h = relu(agg) @ W3 (C=64, no output ReLU)
    gemm_bf16_wmma_kernel<OUT_CH, true><<<nBlkGemm, TB, 0, stream>>>(agg, W3, h, N_NODES);
    init_agg_kernel<OUT_CH><<<nBlk64, TB, 0, stream>>>(h, dinv, b3, agg, N_NODES * 64);
    scatter_kernel<<<nBlkScat, TB, 0, stream>>>(src, dst, dinv, h, agg, N_EDGES, OUT_CH);

    // Mean pool + FC head
    zero_kernel<<<(N_GRAPHS * OUT_CH + N_GRAPHS + TB - 1) / TB, TB, 0, stream>>>(
        gsum, N_GRAPHS * OUT_CH + N_GRAPHS);
    pool_accum_kernel<<<nBlk64, TB, 0, stream>>>(agg, batch, gsum, cnt, N_NODES);
    head_kernel<<<1, 128, 0, stream>>>(gsum, cnt, Wfc, bfc, out);
}
